// TEMModel_83107617177740
// MI455X (gfx1250) — matrix operand order repaired and checked by reading, not compiled, not run on hardware
//
#include <hip/hip_runtime.h>
#include <math.h>

// ---------------------------------------------------------------------------
// Types for CDNA5 WMMA (wave32, 16x16x32 bf16 -> f32)
// ---------------------------------------------------------------------------
typedef __bf16 bf16;
typedef bf16  v16bf __attribute__((ext_vector_type(16)));
typedef float v8f   __attribute__((ext_vector_type(8)));

// Problem sizes
constexpr int CB = 512;      // batch
constexpr int CL = 64;       // location dim
constexpr int CS = 128;      // sensory dim
constexpr int CA = 8;        // action dim
constexpr int CE = 256;      // hidden dim
constexpr int CM = 131072;   // memory rows
constexpr int TOPK = 8;

// Tiling
constexpr int BT  = 64;      // batch tile (rows) for both big kernels
constexpr int MT  = 1024;    // memory columns per score block
constexpr int KT  = 64;      // score sub-tile of memory columns
constexpr int NCH = CM / MT; // 128 chunks for top-k phase 1
constexpr int NSPLIT = 32;   // attention M splits
constexpr int ATILE  = 64;   // keys per attention inner tile

// ---------------------------------------------------------------------------
// Workspace layout (float offsets)
// ---------------------------------------------------------------------------
constexpr long W_LP    = 0;
constexpr long W_MEANE = 16;
constexpr long W_SDE   = W_MEANE + (long)CB * CL;
constexpr long W_EMEAN = W_SDE   + (long)CB * CL;
constexpr long W_ESD   = W_EMEAN + (long)CB * CL;
constexpr long W_DMEAN = W_ESD   + (long)CB * CL;
constexpr long W_DSD   = W_DMEAN + (long)CB * CL;
constexpr long W_OMEAN = W_DSD   + (long)CB * CL;
constexpr long W_OSD   = W_OMEAN + (long)CB * CL;
constexpr long W_SUNIT = W_OSD   + (long)CB * CL;          // B*S
constexpr long W_INVN  = W_SUNIT + (long)CB * CS;          // M
constexpr long W_CANDV = W_INVN  + (long)CM;               // B*NCH*TOPK
constexpr long W_CANDI = W_CANDV + (long)CB * NCH * TOPK;  // (int)
constexpr long W_TOPV  = W_CANDI + (long)CB * NCH * TOPK;  // B*TOPK
constexpr long W_TOPI  = W_TOPV  + (long)CB * TOPK;        // (int)
constexpr long W_FOUND = W_TOPI  + (long)CB * TOPK;        // (int) B
constexpr long W_OBJL  = W_FOUND + (long)CB;               // B*L
constexpr long W_OBJG  = W_OBJL  + (long)CB * CL;          // B*L
constexpr long W_OPART = W_OBJG  + (long)CB * CL;          // NSPLIT*B*S
constexpr long W_MPART = W_OPART + (long)NSPLIT * CB * CS; // NSPLIT*B
constexpr long W_LPART = W_MPART + (long)NSPLIT * CB;      // NSPLIT*B
constexpr long W_SMEAN = W_LPART + (long)NSPLIT * CB;      // B*S
constexpr long W_SSD   = W_SMEAN + (long)CB * CS;          // B*S

// ---------------------------------------------------------------------------
// WMMA fragment helpers (CDNA5 ISA 7.12.2 layouts, wave32)
// ---------------------------------------------------------------------------
__device__ inline v16bf frag_a(const bf16* sm, int ld, int lane) {
  // A 16x32 (MxK): lanes 0-15 row m hold K {0..7,16..23}; lanes 16-31 hold {8..15,24..31}
  const int half = lane >> 4, m = lane & 15;
  v16bf f;
#pragma unroll
  for (int e = 0; e < 16; ++e) {
    const int g = e >> 3, j = e & 7;
    f[e] = sm[m * ld + g * 16 + half * 8 + j];
  }
  return f;
}

__device__ inline v16bf frag_bt(const bf16* sm, int ld, int lane) {
  // B 32x16 (KxN) sourced from a row-major [n][k] tile (i.e. B^T in LDS)
  const int half = lane >> 4, n = lane & 15;
  v16bf f;
#pragma unroll
  for (int e = 0; e < 16; ++e) f[e] = sm[n * ld + (e >> 3) * 16 + half * 8 + (e & 7)];
  return f;
}

__device__ inline v8f wmma_bf(v16bf a, v16bf b, v8f c) {
  return __builtin_amdgcn_wmma_f32_16x16x32_bf16(false, a, false, b, (short)0, c,
                                                 false, false);
}

// ---------------------------------------------------------------------------
// Small utility kernels
// ---------------------------------------------------------------------------
__global__ void init_lp(float* lp) { lp[0] = 0.f; }
__global__ void write_lp(const float* lp, float* out) { out[0] = lp[0]; }

// x = mean + noise*sd; accumulate gaussian log-prob into lp (mean over batch)
__global__ __launch_bounds__(256) void sample_lp(
    const float* __restrict__ mean, const float* __restrict__ sd,
    const float* __restrict__ noise, float* __restrict__ xout,
    int n, float* __restrict__ lp)
{
  __shared__ float red[256];
  const int i = blockIdx.x * 256 + threadIdx.x;
  float c = 0.f;
  if (i < n) {
    const float m = mean[i], s = sd[i], z = noise[i];
    const float x = m + z * s;
    xout[i] = x;
    const float zz = (x - m) / s;
    c = -0.5f * zz * zz - __logf(s);
  }
  red[threadIdx.x] = c;
  __syncthreads();
  for (int st = 128; st > 0; st >>= 1) {
    if (threadIdx.x < st) red[threadIdx.x] += red[threadIdx.x + st];
    __syncthreads();
  }
  if (threadIdx.x == 0) atomicAdd(lp, red[0] * (1.f / (float)CB));
}

// row-normalize sensory (B x S)
__global__ __launch_bounds__(128) void unit_rows(const float* __restrict__ x,
                                                 float* __restrict__ out)
{
  const int b = blockIdx.x, t = threadIdx.x;
  __shared__ float red[CS];
  const float v = x[(long)b * CS + t];
  red[t] = v * v;
  __syncthreads();
  for (int st = 64; st > 0; st >>= 1) { if (t < st) red[t] += red[t + st]; __syncthreads(); }
  out[(long)b * CS + t] = v / (sqrtf(red[0]) + 1e-8f);
}

// inverse row norms of mem_vals (M x S)
__global__ __launch_bounds__(128) void row_inv_norm(const float* __restrict__ vals,
                                                    float* __restrict__ invn)
{
  const int m = blockIdx.x, t = threadIdx.x;
  __shared__ float red[CS];
  const float v = vals[(long)m * CS + t];
  red[t] = v * v;
  __syncthreads();
  for (int st = 64; st > 0; st >>= 1) { if (t < st) red[t] += red[t + st]; __syncthreads(); }
  if (t == 0) invn[m] = 1.f / (sqrtf(red[0]) + 1e-8f);
}

// ---------------------------------------------------------------------------
// Localizer MLP: mean = relu([loc,act]@W1+b1)@W2+b2; sd = exp(scale)*exp(v)+1e-6
// One block (256 threads) per batch row.
// ---------------------------------------------------------------------------
__global__ __launch_bounds__(256) void localizer(
    const float* __restrict__ locin, const float* __restrict__ actin,
    const float* __restrict__ W1, const float* __restrict__ b1,
    const float* __restrict__ W2, const float* __restrict__ b2,
    const float* __restrict__ V1, const float* __restrict__ c1,
    const float* __restrict__ V2, const float* __restrict__ c2,
    const float* __restrict__ scale,
    float* __restrict__ mean_out, float* __restrict__ sd_out)
{
  const int b = blockIdx.x, tid = threadIdx.x;
  __shared__ float xs[CL + CA];
  __shared__ float hs[CE];
  __shared__ float ms[CL];
  if (tid < CL) xs[tid] = locin[(long)b * CL + tid];
  else if (tid < CL + CA) xs[tid] = actin[(long)b * CA + (tid - CL)];
  __syncthreads();
  float h = b1[tid];
  for (int i = 0; i < CL + CA; ++i) h += xs[i] * W1[i * CE + tid];
  hs[tid] = fmaxf(h, 0.f);
  __syncthreads();
  if (tid < CL) {
    float m = b2[tid];
    for (int e = 0; e < CE; ++e) m += hs[e] * W2[e * CL + tid];
    ms[tid] = m;
    mean_out[(long)b * CL + tid] = m;
  }
  __syncthreads();
  float v1 = c1[tid];
  for (int l = 0; l < CL; ++l) v1 += ms[l] * V1[l * CE + tid];
  hs[tid] = fmaxf(v1, 0.f);
  __syncthreads();
  if (tid < CL) {
    float v = c2[tid];
    for (int e = 0; e < CE; ++e) v += hs[e] * V2[e * CL + tid];
    sd_out[(long)b * CL + tid] = __expf(scale[tid]) * __expf(v) + 1e-6f;
  }
}

// ---------------------------------------------------------------------------
// Encoder: scores = unit(sensory) @ unit(mem_vals)^T, fused per-row top-8.
// Block = 64 batch rows x 1024 memory cols, WMMA bf16 16x16x32, K=128.
// ---------------------------------------------------------------------------
__global__ __launch_bounds__(256) void enc_score_topk(
    const float* __restrict__ sunit, const float* __restrict__ mvals,
    const float* __restrict__ invn,
    float* __restrict__ candv, int* __restrict__ candi)
{
  __shared__ bf16  As[BT][CS + 8];   // 64x136 bf16
  __shared__ bf16  Bs[KT][CS + 8];   // 64x136 bf16
  __shared__ float Ss[BT][KT + 1];   // 64x65 f32 (odd stride: bank-conflict-free scans)
  const int tid = threadIdx.x, lane = tid & 31, wave = tid >> 5;
  const int b0 = blockIdx.y * BT;
  const int m0 = blockIdx.x * MT;

  for (int i = tid; i < BT * CS / 4; i += 256) {
    const int r = i / (CS / 4), c = (i % (CS / 4)) * 4;
    const float4 f = *(const float4*)(sunit + (long)(b0 + r) * CS + c);
    As[r][c] = (bf16)f.x; As[r][c + 1] = (bf16)f.y;
    As[r][c + 2] = (bf16)f.z; As[r][c + 3] = (bf16)f.w;
  }
  float tv[TOPK]; int ti[TOPK];
#pragma unroll
  for (int j = 0; j < TOPK; ++j) { tv[j] = -3.0e38f; ti[j] = 0; }

  for (int sub = 0; sub < MT / KT; ++sub) {
    const int mb = m0 + sub * KT;
    __syncthreads();
    if (sub + 1 < MT / KT) __builtin_prefetch(mvals + (long)(mb + KT) * CS, 0, 0);
    for (int i = tid; i < KT * CS / 4; i += 256) {
      const int r = i / (CS / 4), c = (i % (CS / 4)) * 4;
      const float4 f = *(const float4*)(mvals + (long)(mb + r) * CS + c);
      Bs[r][c] = (bf16)f.x; Bs[r][c + 1] = (bf16)f.y;
      Bs[r][c + 2] = (bf16)f.z; Bs[r][c + 3] = (bf16)f.w;
    }
    __syncthreads();
#pragma unroll
    for (int t = 0; t < 2; ++t) {
      const int idx = wave * 2 + t, tr = idx >> 2, tc = idx & 3;
      v8f c = {0.f, 0.f, 0.f, 0.f, 0.f, 0.f, 0.f, 0.f};
#pragma unroll
      for (int kk = 0; kk < CS; kk += 32) {
        const v16bf a = frag_a(&As[tr * 16][0] + kk, CS + 8, lane);
        const v16bf b = frag_bt(&Bs[tc * 16][0] + kk, CS + 8, lane);
        c = wmma_bf(a, b, c);
      }
      const int half = lane >> 4, n = lane & 15;
      const float sc = invn[mb + tc * 16 + n];
#pragma unroll
      for (int r = 0; r < 8; ++r)
        Ss[tr * 16 + r + 8 * half][tc * 16 + n] = c[r] * sc;
    }
    __syncthreads();
    if (tid < BT) {
      for (int j = 0; j < KT; ++j) {
        const float v = Ss[tid][j];
        if (v > tv[TOPK - 1]) {
          int p = TOPK - 1;
          while (p > 0 && v > tv[p - 1]) { tv[p] = tv[p - 1]; ti[p] = ti[p - 1]; --p; }
          tv[p] = v; ti[p] = mb + j;
        }
      }
    }
  }
  if (tid < BT) {
    const long base = ((long)(b0 + tid) * NCH + blockIdx.x) * TOPK;
#pragma unroll
    for (int j = 0; j < TOPK; ++j) { candv[base + j] = tv[j]; candi[base + j] = ti[j]; }
  }
}

// merge per-chunk top-8 candidates -> global top-8 per batch row
__global__ __launch_bounds__(256) void topk_reduce(
    const float* __restrict__ candv, const int* __restrict__ candi,
    float* __restrict__ topv, int* __restrict__ topi, int* __restrict__ found)
{
  const int b = blockIdx.x, tid = threadIdx.x;
  constexpr int NC = NCH * TOPK;  // 1024
  __shared__ float sv[NC];
  __shared__ int   si[NC];
  __shared__ float rv[256];
  __shared__ int   rp[256];
  for (int i = tid; i < NC; i += 256) {
    sv[i] = candv[(long)b * NC + i];
    si[i] = candi[(long)b * NC + i];
  }
  __syncthreads();
  for (int k = 0; k < TOPK; ++k) {
    float bm = -3.0e38f; int bp = 0;
    for (int i = tid; i < NC; i += 256) if (sv[i] > bm) { bm = sv[i]; bp = i; }
    rv[tid] = bm; rp[tid] = bp;
    __syncthreads();
    for (int st = 128; st > 0; st >>= 1) {
      if (tid < st && rv[tid + st] > rv[tid]) { rv[tid] = rv[tid + st]; rp[tid] = rp[tid + st]; }
      __syncthreads();
    }
    if (tid == 0) {
      const int p = rp[0];
      topv[b * TOPK + k] = sv[p];
      topi[b * TOPK + k] = si[p];
      sv[p] = -3.0e38f;
    }
    __syncthreads();
  }
  if (tid == 0) {
    int c = 0;
    for (int k = 0; k < TOPK; ++k) if (topv[b * TOPK + k] > 0.f) ++c;
    found[b] = c;
  }
}

// gather guesses, softmax over valid matches, select obj_location, lp_obj
__global__ __launch_bounds__(64) void enc_select(
    const float* __restrict__ emean, const float* __restrict__ esd,
    const float* __restrict__ mem_keys, const int* __restrict__ topi,
    const int* __restrict__ found, const float* __restrict__ noise_sel,
    float* __restrict__ obj_loc, float* __restrict__ lp)
{
  const int b = blockIdx.x, l = threadIdx.x;
  __shared__ float g[TOPK][CL];
  __shared__ float red[CL];
  __shared__ float match[TOPK];
  __shared__ float attn[TOPK];
  const float em = emean[(long)b * CL + l];
  const float es = esd[(long)b * CL + l];
#pragma unroll
  for (int k = 0; k < TOPK; ++k)
    g[k][l] = mem_keys[(long)topi[b * TOPK + k] * CL + l];
  red[l] = em * em;
  __syncthreads();
  for (int st = 32; st > 0; st >>= 1) { if (l < st) red[l] += red[l + st]; __syncthreads(); }
  const float uem = 1.f / (sqrtf(red[0]) + 1e-8f);
  __syncthreads();
  for (int k = 0; k < TOPK; ++k) {
    red[l] = g[k][l] * g[k][l];
    __syncthreads();
    for (int st = 32; st > 0; st >>= 1) { if (l < st) red[l] += red[l + st]; __syncthreads(); }
    const float gn = sqrtf(red[0]);
    __syncthreads();
    red[l] = g[k][l] * em;
    __syncthreads();
    for (int st = 32; st > 0; st >>= 1) { if (l < st) red[l] += red[l + st]; __syncthreads(); }
    if (l == 0) match[k] = red[0] * uem / (gn + 1e-8f);
    __syncthreads();
  }
  const int fnd = found[b];
  if (l == 0) {
    float mx = -1e30f;
    for (int k = 0; k < TOPK; ++k) {
      const float mv = (k < fnd) ? match[k] : -1e9f;
      match[k] = mv; mx = fmaxf(mx, mv);
    }
    float s = 0.f;
    for (int k = 0; k < TOPK; ++k) { attn[k] = __expf(match[k] - mx); s += attn[k]; }
    for (int k = 0; k < TOPK; ++k) attn[k] /= s;
  }
  __syncthreads();
  float integ = 0.f;
#pragma unroll
  for (int k = 0; k < TOPK; ++k) integ += attn[k] * g[k][l];
  const float d = -0.5f * (integ - em) / es;
  red[l] = d * d;
  __syncthreads();
  for (int st = 32; st > 0; st >>= 1) { if (l < st) red[l] += red[l + st]; __syncthreads(); }
  const bool too_far = sqrtf(red[0]) > 2.f;
  __syncthreads();
  const float expct = em + noise_sel[(long)b * CL + l] * es;
  const float ol = (too_far || (fnd == 0)) ? expct : integ;
  obj_loc[(long)b * CL + l] = ol;
  const float z = (ol - em) / es;
  red[l] = -0.5f * z * z - __logf(es);
  __syncthreads();
  for (int st = 32; st > 0; st >>= 1) { if (l < st) red[l] += red[l + st]; __syncthreads(); }
  if (l == 0) atomicAdd(lp, red[0] * (1.f / (float)CB));
}

// ---------------------------------------------------------------------------
// Decoder attention, flash-style with M split into NSPLIT partials.
// Q=obj_gen (B x 64), K (M x 64), V (M x 128). WMMA bf16, f32 accum.
// V tile is stored transposed in LDS so both GEMMs use contiguous fragments.
// ---------------------------------------------------------------------------
__global__ __launch_bounds__(256) void dec_attn_partial(
    const float* __restrict__ q, const float* __restrict__ keys,
    const float* __restrict__ vals,
    float* __restrict__ Opart, float* __restrict__ mpart, float* __restrict__ lpart)
{
  __shared__ bf16  Qs[BT][CL + 8];      // 64x72
  __shared__ bf16  Ks[ATILE][CL + 8];   // 64x72
  __shared__ float Ss[BT][ATILE + 1];   // 64x65 (odd stride)
  __shared__ bf16  Ps[BT][ATILE + 8];   // 64x72
  __shared__ bf16  Vt[CS][ATILE + 8];   // 128x72 (transposed: [s][k])
  __shared__ float rowm[BT], rowl[BT], ralpha[BT];
  const int tid = threadIdx.x, lane = tid & 31, wave = tid >> 5;
  const int split = blockIdx.x, b0 = blockIdx.y * BT;
  const int mbeg = split * (CM / NSPLIT);

  for (int i = tid; i < BT * CL / 4; i += 256) {
    const int r = i / (CL / 4), c = (i % (CL / 4)) * 4;
    const float4 f = *(const float4*)(q + (long)(b0 + r) * CL + c);
    Qs[r][c] = (bf16)f.x; Qs[r][c + 1] = (bf16)f.y;
    Qs[r][c + 2] = (bf16)f.z; Qs[r][c + 3] = (bf16)f.w;
  }
  if (tid < BT) { rowm[tid] = -3.0e38f; rowl[tid] = 0.f; }
  const v8f vzero = {0.f, 0.f, 0.f, 0.f, 0.f, 0.f, 0.f, 0.f};
  v8f o[4];
#pragma unroll
  for (int t = 0; t < 4; ++t) o[t] = vzero;
  __syncthreads();

  for (int kt = 0; kt < CM / NSPLIT; kt += ATILE) {
    const int mb = mbeg + kt;
    __builtin_prefetch(keys + (long)(mb + ATILE) * CL, 0, 0);
    __builtin_prefetch(vals + (long)(mb + ATILE) * CS, 0, 0);
    for (int i = tid; i < ATILE * CL / 4; i += 256) {
      const int r = i / (CL / 4), c = (i % (CL / 4)) * 4;
      const float4 f = *(const float4*)(keys + (long)(mb + r) * CL + c);
      Ks[r][c] = (bf16)f.x; Ks[r][c + 1] = (bf16)f.y;
      Ks[r][c + 2] = (bf16)f.z; Ks[r][c + 3] = (bf16)f.w;
    }
    for (int i = tid; i < ATILE * CS / 4; i += 256) {
      const int r = i / (CS / 4), c = (i % (CS / 4)) * 4;   // r = key row, c = s col
      const float4 f = *(const float4*)(vals + (long)(mb + r) * CS + c);
      Vt[c + 0][r] = (bf16)f.x; Vt[c + 1][r] = (bf16)f.y;
      Vt[c + 2][r] = (bf16)f.z; Vt[c + 3][r] = (bf16)f.w;
    }
    __syncthreads();
    // S = Q K^T / sqrt(L) : 64x64 -> 16 tiles, 2 per wave
#pragma unroll
    for (int t = 0; t < 2; ++t) {
      const int idx = wave * 2 + t, tr = idx >> 2, tc = idx & 3;
      v8f c = vzero;
#pragma unroll
      for (int kk = 0; kk < CL; kk += 32) {
        const v16bf a = frag_a(&Qs[tr * 16][0] + kk, CL + 8, lane);
        const v16bf bb = frag_bt(&Ks[tc * 16][0] + kk, CL + 8, lane);
        c = wmma_bf(a, bb, c);
      }
      const int half = lane >> 4, n = lane & 15;
#pragma unroll
      for (int r = 0; r < 8; ++r)
        Ss[tr * 16 + r + 8 * half][tc * 16 + n] = c[r] * 0.125f;
    }
    __syncthreads();
    // online softmax row stats
    if (tid < BT) {
      float mx = rowm[tid];
      for (int j = 0; j < ATILE; ++j) mx = fmaxf(mx, Ss[tid][j]);
      const float al = __expf(rowm[tid] - mx);
      float sum = 0.f;
      for (int j = 0; j < ATILE; ++j) {
        const float p = __expf(Ss[tid][j] - mx);
        Ss[tid][j] = p; sum += p;
      }
      ralpha[tid] = al; rowm[tid] = mx;
      rowl[tid] = rowl[tid] * al + sum;
    }
    __syncthreads();
    for (int i = tid; i < BT * ATILE; i += 256) {
      const int r = i >> 6, c = i & 63;
      Ps[r][c] = (bf16)Ss[r][c];
    }
    __syncthreads();
    // O = O*alpha + P @ V : 64x128 -> 32 tiles, 4 per wave
#pragma unroll
    for (int t = 0; t < 4; ++t) {
      const int idx = wave * 4 + t, tr = idx >> 3, tc = idx & 7;
      const int half = lane >> 4;
      v8f c = o[t];
#pragma unroll
      for (int r = 0; r < 8; ++r) c[r] *= ralpha[tr * 16 + r + 8 * half];
#pragma unroll
      for (int kk = 0; kk < ATILE; kk += 32) {
        const v16bf a = frag_a(&Ps[tr * 16][0] + kk, ATILE + 8, lane);
        const v16bf bb = frag_bt(&Vt[tc * 16][0] + kk, ATILE + 8, lane);
        c = wmma_bf(a, bb, c);
      }
      o[t] = c;
    }
    __syncthreads();
  }
#pragma unroll
  for (int t = 0; t < 4; ++t) {
    const int idx = wave * 4 + t, tr = idx >> 3, tc = idx & 7;
    const int half = lane >> 4, n = lane & 15;
#pragma unroll
    for (int r = 0; r < 8; ++r) {
      const int brow = b0 + tr * 16 + r + 8 * half;
      Opart[((long)split * CB + brow) * CS + tc * 16 + n] = o[t][r];
    }
  }
  if (tid < BT) {
    mpart[(long)split * CB + b0 + tid] = rowm[tid];
    lpart[(long)split * CB + b0 + tid] = rowl[tid];
  }
}

// merge attention partials with log-sum-exp
__global__ __launch_bounds__(128) void attn_combine(
    const float* __restrict__ Opart, const float* __restrict__ mpart,
    const float* __restrict__ lpart, float* __restrict__ smean)
{
  const int b = blockIdx.x, n = threadIdx.x;
  float gm = -3.0e38f;
  for (int j = 0; j < NSPLIT; ++j) gm = fmaxf(gm, mpart[(long)j * CB + b]);
  float denom = 0.f, acc = 0.f;
  for (int j = 0; j < NSPLIT; ++j) {
    const float w = __expf(mpart[(long)j * CB + b] - gm);
    denom += lpart[(long)j * CB + b] * w;
    acc += Opart[((long)j * CB + b) * CS + n] * w;
  }
  smean[(long)b * CS + n] = acc / denom;
}

// decoder sensory sd MLP: sd = exp(svar)*exp(relu([nlg,objg]@sW1+sb1)@sW2+sb2)+1e-6
__global__ __launch_bounds__(256) void dec_sensory_sd(
    const float* __restrict__ nlg, const float* __restrict__ objg,
    const float* __restrict__ sW1, const float* __restrict__ sb1,
    const float* __restrict__ sW2, const float* __restrict__ sb2,
    const float* __restrict__ svar, float* __restrict__ sd_out)
{
  const int b = blockIdx.x, tid = threadIdx.x;
  __shared__ float xs[2 * CL];
  __shared__ float hs[CE];
  if (tid < CL) xs[tid] = nlg[(long)b * CL + tid];
  else if (tid < 2 * CL) xs[tid] = objg[(long)b * CL + (tid - CL)];
  __syncthreads();
  float h = sb1[tid];
  for (int i = 0; i < 2 * CL; ++i) h += xs[i] * sW1[i * CE + tid];
  hs[tid] = fmaxf(h, 0.f);
  __syncthreads();
  if (tid < CS) {
    float v = sb2[tid];
    for (int e = 0; e < CE; ++e) v += hs[e] * sW2[e * CS + tid];
    sd_out[(long)b * CS + tid] = __expf(svar[tid]) * __expf(v) + 1e-6f;
  }
}

// copy M x cols table replacing the first 512 rows (non-temporal streaming)
__global__ void copy_replace(float* __restrict__ dst, const float* __restrict__ src,
                             const float* __restrict__ repl, long n, int cols)
{
  const long stride = (long)gridDim.x * blockDim.x;
  for (long i = (long)blockIdx.x * blockDim.x + threadIdx.x; i < n; i += stride) {
    const long row = i / cols;
    const float v = (row < CB) ? repl[i] : __builtin_nontemporal_load(src + i);
    __builtin_nontemporal_store(v, dst + i);
  }
}

// ---------------------------------------------------------------------------
// Launch
// ---------------------------------------------------------------------------
extern "C" void kernel_launch(void* const* d_in, const int* in_sizes, int n_in,
                              void* d_out, int out_size, void* d_ws, size_t ws_size,
                              hipStream_t stream)
{
  (void)in_sizes; (void)n_in; (void)out_size; (void)ws_size;
  const float* sensory  = (const float*)d_in[0];
  const float* action   = (const float*)d_in[1];
  const float* last_loc = (const float*)d_in[2];
  const float* heading  = (const float*)d_in[3];
  const float* eW1 = (const float*)d_in[4];
  const float* eb1 = (const float*)d_in[5];
  const float* eW2 = (const float*)d_in[6];
  const float* eb2 = (const float*)d_in[7];
  const float* eV1 = (const float*)d_in[8];
  const float* ec1 = (const float*)d_in[9];
  const float* eV2 = (const float*)d_in[10];
  const float* ec2 = (const float*)d_in[11];
  const float* escale = (const float*)d_in[12];
  const float* emem_k = (const float*)d_in[13];
  const float* emem_v = (const float*)d_in[14];
  const float* dW1 = (const float*)d_in[15];
  const float* db1 = (const float*)d_in[16];
  const float* dW2 = (const float*)d_in[17];
  const float* db2 = (const float*)d_in[18];
  const float* dV1 = (const float*)d_in[19];
  const float* dc1 = (const float*)d_in[20];
  const float* dV2 = (const float*)d_in[21];
  const float* dc2 = (const float*)d_in[22];
  const float* dscale = (const float*)d_in[23];
  const float* dmem_k = (const float*)d_in[24];
  const float* dmem_v = (const float*)d_in[25];
  const float* sW1 = (const float*)d_in[26];
  const float* sb1 = (const float*)d_in[27];
  const float* sW2 = (const float*)d_in[28];
  const float* sb2 = (const float*)d_in[29];
  const float* svar = (const float*)d_in[30];
  const float* n_enc_loc = (const float*)d_in[31];
  const float* n_enc_sel = (const float*)d_in[32];
  const float* n_dec_loc = (const float*)d_in[33];
  const float* n_dec_obj = (const float*)d_in[34];
  const float* n_sens    = (const float*)d_in[35];

  float* out = (float*)d_out;
  float* ws  = (float*)d_ws;

  float* lp      = ws + W_LP;
  float* mean_e  = ws + W_MEANE;
  float* sd_e    = ws + W_SDE;
  float* emean_  = ws + W_EMEAN;
  float* esd_    = ws + W_ESD;
  float* dmean_  = ws + W_DMEAN;
  float* dsd_    = ws + W_DSD;
  float* omean_  = ws + W_OMEAN;
  float* osd_    = ws + W_OSD;
  float* sunit   = ws + W_SUNIT;
  float* invn    = ws + W_INVN;
  float* candv   = ws + W_CANDV;
  int*   candi   = (int*)(ws + W_CANDI);
  float* topv    = ws + W_TOPV;
  int*   topi_p  = (int*)(ws + W_TOPI);
  int*   found   = (int*)(ws + W_FOUND);
  float* obj_loc = ws + W_OBJL;
  float* obj_gen = ws + W_OBJG;
  float* Opart   = ws + W_OPART;
  float* mpart   = ws + W_MPART;
  float* lpart   = ws + W_LPART;
  float* smean   = ws + W_SMEAN;
  float* ssd     = ws + W_SSD;

  float* o_nextloc = out;                               // B*L
  float* o_nlg     = out + (long)CB * CL;               // B*L
  float* o_sensgen = out + 2L * CB * CL;                // B*S
  float* o_lp      = out + 2L * CB * CL + (long)CB * CS;          // 1
  float* o_ekeys   = o_lp + 1;                          // M*L
  float* o_evals   = o_ekeys + (long)CM * CL;           // M*S
  float* o_dkeys   = o_evals + (long)CM * CS;           // M*L
  float* o_dvals   = o_dkeys + (long)CM * CL;           // M*S

  init_lp<<<1, 1, 0, stream>>>(lp);

  // encoder localizer -> next_location + lp_loc
  localizer<<<CB, 256, 0, stream>>>(last_loc, action, eW1, eb1, eW2, eb2,
                                    eV1, ec1, eV2, ec2, escale, mean_e, sd_e);
  sample_lp<<<(CB * CL) / 256, 256, 0, stream>>>(mean_e, sd_e, n_enc_loc,
                                                 o_nextloc, CB * CL, lp);
  localizer<<<CB, 256, 0, stream>>>(o_nextloc, heading, eW1, eb1, eW2, eb2,
                                    eV1, ec1, eV2, ec2, escale, emean_, esd_);

  // encoder memory search (WMMA) + top-k + select
  unit_rows<<<CB, 128, 0, stream>>>(sensory, sunit);
  row_inv_norm<<<CM, 128, 0, stream>>>(emem_v, invn);
  dim3 gscore(CM / MT, CB / BT);   // (128, 8)
  enc_score_topk<<<gscore, 256, 0, stream>>>(sunit, emem_v, invn, candv, candi);
  topk_reduce<<<CB, 256, 0, stream>>>(candv, candi, topv, topi_p, found);
  enc_select<<<CB, 64, 0, stream>>>(emean_, esd_, emem_k, topi_p, found,
                                    n_enc_sel, obj_loc, lp);

  // decoder generative path
  localizer<<<CB, 256, 0, stream>>>(last_loc, action, dW1, db1, dW2, db2,
                                    dV1, dc1, dV2, dc2, dscale, dmean_, dsd_);
  sample_lp<<<(CB * CL) / 256, 256, 0, stream>>>(dmean_, dsd_, n_dec_loc,
                                                 o_nlg, CB * CL, lp);
  localizer<<<CB, 256, 0, stream>>>(o_nlg, heading, dW1, db1, dW2, db2,
                                    dV1, dc1, dV2, dc2, dscale, omean_, osd_);
  sample_lp<<<(CB * CL) / 256, 256, 0, stream>>>(omean_, osd_, n_dec_obj,
                                                 obj_gen, CB * CL, lp);

  // decoder associative read (flash attention, WMMA)
  dim3 gattn(NSPLIT, CB / BT);     // (32, 8)
  dec_attn_partial<<<gattn, 256, 0, stream>>>(obj_gen, dmem_k, dmem_v,
                                              Opart, mpart, lpart);
  attn_combine<<<CB, 128, 0, stream>>>(Opart, mpart, lpart, smean);
  dec_sensory_sd<<<CB, 256, 0, stream>>>(o_nlg, obj_gen, sW1, sb1, sW2, sb2,
                                         svar, ssd);
  sample_lp<<<(CB * CS) / 256, 256, 0, stream>>>(smean, ssd, n_sens,
                                                 o_sensgen, CB * CS, lp);

  // ring-buffer memory writes (copy + row replace)
  copy_replace<<<2048, 256, 0, stream>>>(o_ekeys, emem_k, obj_loc, (long)CM * CL, CL);
  copy_replace<<<2048, 256, 0, stream>>>(o_evals, emem_v, sensory, (long)CM * CS, CS);
  copy_replace<<<2048, 256, 0, stream>>>(o_dkeys, dmem_k, obj_gen, (long)CM * CL, CL);
  copy_replace<<<2048, 256, 0, stream>>>(o_dvals, dmem_v, sensory, (long)CM * CS, CS);

  write_lp<<<1, 1, 0, stream>>>(lp, o_lp);
}
